// TransformerBlock_18829136626133
// MI455X (gfx1250) — matrix operand order repaired and checked by reading, not compiled, not run on hardware
//
#include <hip/hip_runtime.h>

// ---------------------------------------------------------------------------
// CDNA5 (gfx1250) PointTransformerConv block, wave32 WMMA bf16 pipeline.
// Weights pre-packed to bf16 B-fragment layout; LDS tiles staged in bf16 so
// A-fragments are 2x ds_read_b128; global gathers use b128 loads.
// ---------------------------------------------------------------------------

typedef __attribute__((ext_vector_type(16))) __bf16 v16bf;
typedef __attribute__((ext_vector_type(8)))  __bf16 v8bf;
typedef __attribute__((ext_vector_type(8)))  float  v8f;

__device__ __forceinline__ v8f wmma_bf16(v16bf a, v16bf b, v8f c) {
    // (neg_a, A, neg_b, B, c_mod, C, reuse_a, reuse_b)
    return __builtin_amdgcn_wmma_f32_16x16x32_bf16(false, a, false, b, (short)0, c, false, false);
}

// A-fragment (16x32 bf16) from a bf16 LDS tile [16][ldr], K-chunk kc.
// lanes 0-15: rows M=0..15, K = kc*32 + {0..7, 16..23}
// lanes 16-31: K = kc*32 + {8..15, 24..31}
// Per lane: two contiguous 8-element (16B) runs -> 2x ds_read_b128.
__device__ __forceinline__ v16bf load_A_lds(const __bf16* tile, int ldr, int kc) {
    int lane = threadIdx.x & 31;
    int row  = lane & 15;
    int koff = kc * 32 + ((lane & 16) ? 8 : 0);
    const __bf16* p = tile + row * ldr + koff;
    v8bf lo = *(const v8bf*)p;
    v8bf hi = *(const v8bf*)(p + 16);
    return __builtin_shufflevector(lo, hi, 0, 1, 2, 3, 4, 5, 6, 7,
                                           8, 9, 10, 11, 12, 13, 14, 15);
}

// B-fragment from pre-packed table: frag f, lane holds 16 contiguous bf16.
__device__ __forceinline__ v16bf load_B_frag(const __bf16* __restrict__ Wp, int frag) {
    int lane = threadIdx.x & 31;
    const __bf16* p = Wp + (frag * 32 + lane) * 16;
    v8bf lo = *(const v8bf*)p;
    v8bf hi = *(const v8bf*)(p + 8);
    return __builtin_shufflevector(lo, hi, 0, 1, 2, 3, 4, 5, 6, 7,
                                           8, 9, 10, 11, 12, 13, 14, 15);
}

__device__ __forceinline__ v8f splat_frag(float bv) {
    v8f c;
#pragma unroll
    for (int r = 0; r < 8; ++r) c[r] = bv;
    return c;
}

__device__ __forceinline__ v8f zero_frag() {
    v8f c;
#pragma unroll
    for (int r = 0; r < 8; ++r) c[r] = 0.0f;
    return c;
}

__device__ __forceinline__ v8f relu_frag(v8f c) {
#pragma unroll
    for (int r = 0; r < 8; ++r) c[r] = fmaxf(c[r], 0.0f);
    return c;
}

// C layout: VGPR r, lanes 0-15: (M=r, N=lane); lanes 16-31: (M=r+8, N=lane-16)
__device__ __forceinline__ void store_C_lds_bf16(__bf16* tile, int ldr, int nt, v8f c) {
    int lane  = threadIdx.x & 31;
    int mbase = (lane & 16) ? 8 : 0;
    int n     = nt * 16 + (lane & 15);
#pragma unroll
    for (int r = 0; r < 8; ++r) tile[(mbase + r) * ldr + n] = (__bf16)c[r];
}

__device__ __forceinline__ void store_C_global(float* __restrict__ out, long long row0, int nt, v8f c) {
    int lane  = threadIdx.x & 31;
    int mbase = (lane & 16) ? 8 : 0;
    int n     = nt * 16 + (lane & 15);
#pragma unroll
    for (int r = 0; r < 8; ++r) out[(row0 + mbase + r) * 64 + n] = c[r];
}

__device__ __forceinline__ v8bf cvt8(float4 a, float4 b) {
    v8bf r;
    r[0] = (__bf16)a.x; r[1] = (__bf16)a.y; r[2] = (__bf16)a.z; r[3] = (__bf16)a.w;
    r[4] = (__bf16)b.x; r[5] = (__bf16)b.y; r[6] = (__bf16)b.z; r[7] = (__bf16)b.w;
    return r;
}

// ---------------------------------------------------------------------------
// Kernel P: pack 8x (64x64) weights + pos_w1 (64x3, K padded to 32) into
// bf16 B-fragment tables: table[frag][lane][e], frag = nt*2+kc (or nt).
// ---------------------------------------------------------------------------
__global__ void k_pack_weights(const float* __restrict__ W0, const float* __restrict__ W1,
                               const float* __restrict__ W2, const float* __restrict__ W3,
                               const float* __restrict__ W4, const float* __restrict__ W5,
                               const float* __restrict__ W6, const float* __restrict__ W7,
                               const float* __restrict__ P1, __bf16* __restrict__ out) {
    int t = blockIdx.x * blockDim.x + threadIdx.x;
    if (t < 8 * 4096) {
        int w = t >> 12, r = t & 4095;
        int e = r & 15, lane = (r >> 4) & 31, kc = (r >> 9) & 1, nt = r >> 10;
        const float* W = (w == 0) ? W0 : (w == 1) ? W1 : (w == 2) ? W2 : (w == 3) ? W3
                        : (w == 4) ? W4 : (w == 5) ? W5 : (w == 6) ? W6 : W7;
        out[t] = (__bf16)W[(nt * 16 + (lane & 15)) * 64 + kc * 32 + ((e & 8) << 1) + ((lane & 16) >> 1) + (e & 7)];
    } else if (t < 8 * 4096 + 2048) {
        int r = t - 8 * 4096;
        int e = r & 15, lane = (r >> 4) & 31, nt = r >> 9;
        int kk = ((e & 8) << 1) + ((lane & 16) >> 1) + (e & 7);
        float val = (kk < 3) ? P1[(nt * 16 + (lane & 15)) * 3 + kk] : 0.0f;
        out[t] = (__bf16)val;
    }
}

// ---------------------------------------------------------------------------
// Kernel 0: zero scratch accumulators (float4)
// ---------------------------------------------------------------------------
__global__ void k_zero4(float4* __restrict__ p, long long n4) {
    long long i = (long long)blockIdx.x * blockDim.x + threadIdx.x;
    if (i < n4) p[i] = make_float4(0.f, 0.f, 0.f, 0.f);
}

// ---------------------------------------------------------------------------
// Kernel 1: h = relu(x@W_in^T+b); v = h@W_lin^T; k = h@W_src^T; q = h@W_dst^T
// ---------------------------------------------------------------------------
__global__ void k_node_proj(const float* __restrict__ x,
                            const __bf16* __restrict__ Wp_in, const float* __restrict__ b_in,
                            const __bf16* __restrict__ Wp_lin,
                            const __bf16* __restrict__ Wp_src,
                            const __bf16* __restrict__ Wp_dst,
                            float* __restrict__ v, float* __restrict__ ks, float* __restrict__ qd,
                            int ntiles) {
    __shared__ __align__(16) __bf16 tiles[4][16 * 64];
    int wid  = threadIdx.x >> 5;
    int tile = blockIdx.x * 4 + wid;
    if (tile >= ntiles) return;
    __bf16* t = tiles[wid];
    int lane  = threadIdx.x & 31;
    long long row0 = (long long)tile * 16;

    float bin[4];   // bias values hoisted out of the nt loops
#pragma unroll
    for (int nt = 0; nt < 4; ++nt) bin[nt] = b_in[nt * 16 + (lane & 15)];

    { // stage 16x64 x-tile into bf16 LDS: lane -> (row = lane/2, 32 cols)
        int r  = lane >> 1;
        int c0 = (lane & 1) * 32;
        const float4* src = (const float4*)(x + (row0 + r) * 64 + c0);
        v8bf* dst = (v8bf*)(t + r * 64 + c0);
        dst[0] = cvt8(src[0], src[1]);
        dst[1] = cvt8(src[2], src[3]);
        dst[2] = cvt8(src[4], src[5]);
        dst[3] = cvt8(src[6], src[7]);
    }
    __builtin_amdgcn_wave_barrier();
    v16bf a0 = load_A_lds(t, 64, 0);
    v16bf a1 = load_A_lds(t, 64, 1);
    __builtin_amdgcn_wave_barrier();
#pragma unroll
    for (int nt = 0; nt < 4; ++nt) {   // h tile -> back into same LDS buffer
        v8f c = splat_frag(bin[nt]);
        c = wmma_bf16(a0, load_B_frag(Wp_in, nt * 2 + 0), c);
        c = wmma_bf16(a1, load_B_frag(Wp_in, nt * 2 + 1), c);
        store_C_lds_bf16(t, 64, nt, relu_frag(c));
    }
    __builtin_amdgcn_wave_barrier();
    v16bf h0 = load_A_lds(t, 64, 0);
    v16bf h1 = load_A_lds(t, 64, 1);
#pragma unroll
    for (int nt = 0; nt < 4; ++nt) {
        v8f c = zero_frag();
        c = wmma_bf16(h0, load_B_frag(Wp_lin, nt * 2 + 0), c);
        c = wmma_bf16(h1, load_B_frag(Wp_lin, nt * 2 + 1), c);
        store_C_global(v, row0, nt, c);
    }
#pragma unroll
    for (int nt = 0; nt < 4; ++nt) {
        v8f c = zero_frag();
        c = wmma_bf16(h0, load_B_frag(Wp_src, nt * 2 + 0), c);
        c = wmma_bf16(h1, load_B_frag(Wp_src, nt * 2 + 1), c);
        store_C_global(ks, row0, nt, c);
    }
#pragma unroll
    for (int nt = 0; nt < 4; ++nt) {
        v8f c = zero_frag();
        c = wmma_bf16(h0, load_B_frag(Wp_dst, nt * 2 + 0), c);
        c = wmma_bf16(h1, load_B_frag(Wp_dst, nt * 2 + 1), c);
        store_C_global(qd, row0, nt, c);
    }
}

// ---------------------------------------------------------------------------
// Kernel 2: alpha = relu(relu((q_d - k_s)@w1^T+b1)@w2^T+b2); amax atomicMax
// (alpha >= 0 post-ReLU, amax zero-initialized -> unsigned-bit max is exact)
// ---------------------------------------------------------------------------
__global__ void k_edge_alpha(const int* __restrict__ ei,
                             const float* __restrict__ ks, const float* __restrict__ qd,
                             const __bf16* __restrict__ Wp1, const float* __restrict__ b1,
                             const __bf16* __restrict__ Wp2, const float* __restrict__ b2,
                             float* __restrict__ alpha, float* __restrict__ amax,
                             int E, int ntiles) {
    __shared__ __align__(16) __bf16 tiles[4][16 * 64];
    __shared__ int dsts[4][16];
    int wid  = threadIdx.x >> 5;
    int tile = blockIdx.x * 4 + wid;
    if (tile >= ntiles) return;
    __bf16* t = tiles[wid];
    int*   dA = dsts[wid];
    int lane  = threadIdx.x & 31;
    int e0    = tile * 16;
    __builtin_prefetch(ei + e0 + 64, 0, 1);

    float bv1[4], bv2[4];
#pragma unroll
    for (int nt = 0; nt < 4; ++nt) {
        bv1[nt] = b1[nt * 16 + (lane & 15)];
        bv2[nt] = b2[nt * 16 + (lane & 15)];
    }

    { // gather diff rows: lane -> (row = lane/2, 32 of 64 cols), b128 loads
        int r    = lane >> 1;
        int half = lane & 1;
        int e = e0 + r;
        int s, d;
        if (e < E) { s = ei[e]; d = ei[E + e]; } else { s = e - E; d = s; }
        if (half == 0) dA[r] = d;
        int c0 = half * 32;
        const float4* kp = (const float4*)(ks + (long long)s * 64 + c0);
        const float4* qp = (const float4*)(qd + (long long)d * 64 + c0);
        v8bf* dst = (v8bf*)(t + r * 64 + c0);
#pragma unroll
        for (int i = 0; i < 4; ++i) {
            float4 qa = qp[2 * i], qb = qp[2 * i + 1];
            float4 ka = kp[2 * i], kb = kp[2 * i + 1];
            float4 da = make_float4(qa.x - ka.x, qa.y - ka.y, qa.z - ka.z, qa.w - ka.w);
            float4 db = make_float4(qb.x - kb.x, qb.y - kb.y, qb.z - kb.z, qb.w - kb.w);
            dst[i] = cvt8(da, db);
        }
    }
    __builtin_amdgcn_wave_barrier();
    v16bf a0 = load_A_lds(t, 64, 0);
    v16bf a1 = load_A_lds(t, 64, 1);
    __builtin_amdgcn_wave_barrier();
#pragma unroll
    for (int nt = 0; nt < 4; ++nt) {
        v8f c = splat_frag(bv1[nt]);
        c = wmma_bf16(a0, load_B_frag(Wp1, nt * 2 + 0), c);
        c = wmma_bf16(a1, load_B_frag(Wp1, nt * 2 + 1), c);
        store_C_lds_bf16(t, 64, nt, relu_frag(c));
    }
    __builtin_amdgcn_wave_barrier();
    v16bf y0 = load_A_lds(t, 64, 0);
    v16bf y1 = load_A_lds(t, 64, 1);
    int mbase = (lane & 16) ? 8 : 0;
    int ncol  = lane & 15;
#pragma unroll
    for (int nt = 0; nt < 4; ++nt) {
        v8f c = splat_frag(bv2[nt]);
        c = wmma_bf16(y0, load_B_frag(Wp2, nt * 2 + 0), c);
        c = wmma_bf16(y1, load_B_frag(Wp2, nt * 2 + 1), c);
        c = relu_frag(c);
        int n = nt * 16 + ncol;
#pragma unroll
        for (int r = 0; r < 8; ++r) {
            int m = mbase + r;
            float val = c[r];
            alpha[(long long)(e0 + m) * 64 + n] = val;
            int d = dA[m];
            atomicMax((unsigned int*)(amax + (long long)d * 64 + n), __float_as_uint(val));
        }
    }
}

// ---------------------------------------------------------------------------
// Kernel 3: ex = exp(alpha - amax[dst]); alpha <- ex; denom[dst] += ex
// float4 per thread (4 channels).
// ---------------------------------------------------------------------------
__global__ void k_softmax_norm(const int* __restrict__ ei, const float* __restrict__ amax,
                               float* __restrict__ alpha, float* __restrict__ denom,
                               int E, long long total4) {
    long long idx = (long long)blockIdx.x * blockDim.x + threadIdx.x;
    if (idx >= total4) return;
    long long e = idx >> 4;
    int c0 = (int)(idx & 15) * 4;
    int d = (e < (long long)E) ? ei[E + e] : (int)(e - E);
    float4 a  = *(const float4*)(alpha + e * 64 + c0);
    float4 mx = *(const float4*)(amax + (long long)d * 64 + c0);
    float4 r;
    r.x = __expf(a.x - mx.x);
    r.y = __expf(a.y - mx.y);
    r.z = __expf(a.z - mx.z);
    r.w = __expf(a.w - mx.w);
    *(float4*)(alpha + e * 64 + c0) = r;
    float* dn = denom + (long long)d * 64 + c0;
    atomicAdd(dn + 0, r.x);
    atomicAdd(dn + 1, r.y);
    atomicAdd(dn + 2, r.z);
    atomicAdd(dn + 3, r.w);
}

// ---------------------------------------------------------------------------
// Kernel 3b: denom <- 1/(denom + 1e-16)  (replaces per-edge divide by multiply)
// ---------------------------------------------------------------------------
__global__ void k_inv_denom(float4* __restrict__ denom, long long n4) {
    long long i = (long long)blockIdx.x * blockDim.x + threadIdx.x;
    if (i >= n4) return;
    float4 dv = denom[i];
    dv.x = 1.0f / (dv.x + 1e-16f);
    dv.y = 1.0f / (dv.y + 1e-16f);
    dv.z = 1.0f / (dv.z + 1e-16f);
    dv.w = 1.0f / (dv.w + 1e-16f);
    denom[i] = dv;
}

// ---------------------------------------------------------------------------
// Kernel 4: delta = pos MLP; msg = ex * inv_denom[dst] * (v[src] + delta);
// acc[dst] += msg
// ---------------------------------------------------------------------------
__global__ void k_edge_msg(const int* __restrict__ ei, const float* __restrict__ pos,
                           const float* __restrict__ v,
                           const __bf16* __restrict__ Wp1, const float* __restrict__ pb1,
                           const __bf16* __restrict__ Wp2, const float* __restrict__ pb2,
                           const float* __restrict__ ex, const float* __restrict__ invden,
                           float* __restrict__ acc,
                           int E, int ntiles) {
    __shared__ __align__(16) __bf16 tp[4][16 * 32];
    __shared__ __align__(16) __bf16 ty[4][16 * 64];
    __shared__ int sArr[4][16];
    __shared__ int dArr[4][16];
    int wid  = threadIdx.x >> 5;
    int tile = blockIdx.x * 4 + wid;
    if (tile >= ntiles) return;
    int lane = threadIdx.x & 31;
    __bf16* P = tp[wid];
    __bf16* Y = ty[wid];
    int*   sA = sArr[wid];
    int*   dA = dArr[wid];
    int e0    = tile * 16;
    __builtin_prefetch(ei + e0 + 64, 0, 1);

    float bp1[4], bp2[4];
#pragma unroll
    for (int nt = 0; nt < 4; ++nt) {
        bp1[nt] = pb1[nt * 16 + (lane & 15)];
        bp2[nt] = pb2[nt * 16 + (lane & 15)];
    }

    { // stage pos-diff 16x32 bf16 tile (K = 3, zero padded to 32)
        int r    = lane >> 1;
        int half = lane & 1;
        int e = e0 + r;
        int s, d;
        if (e < E) { s = ei[e]; d = ei[E + e]; } else { s = e - E; d = s; }
        if (half == 0) { sA[r] = s; dA[r] = d; }
        v8bf z;
#pragma unroll
        for (int j = 0; j < 8; ++j) z[j] = (__bf16)0.0f;
        v8bf* dst = (v8bf*)(P + r * 32);
        if (half == 0) {
            v8bf p0 = z;
            p0[0] = (__bf16)(pos[(long long)d * 3 + 0] - pos[(long long)s * 3 + 0]);
            p0[1] = (__bf16)(pos[(long long)d * 3 + 1] - pos[(long long)s * 3 + 1]);
            p0[2] = (__bf16)(pos[(long long)d * 3 + 2] - pos[(long long)s * 3 + 2]);
            dst[0] = p0;
            dst[1] = z;
        } else {
            dst[2] = z;
            dst[3] = z;
        }
    }
    __builtin_amdgcn_wave_barrier();
    v16bf ap = load_A_lds(P, 32, 0);
#pragma unroll
    for (int nt = 0; nt < 4; ++nt) {
        v8f c = splat_frag(bp1[nt]);
        c = wmma_bf16(ap, load_B_frag(Wp1, nt), c);
        store_C_lds_bf16(Y, 64, nt, relu_frag(c));
    }
    __builtin_amdgcn_wave_barrier();
    v16bf y0 = load_A_lds(Y, 64, 0);
    v16bf y1 = load_A_lds(Y, 64, 1);
    int mbase = (lane & 16) ? 8 : 0;
    int ncol  = lane & 15;
#pragma unroll
    for (int nt = 0; nt < 4; ++nt) {
        v8f c = splat_frag(bp2[nt]);
        c = wmma_bf16(y0, load_B_frag(Wp2, nt * 2 + 0), c);
        c = wmma_bf16(y1, load_B_frag(Wp2, nt * 2 + 1), c);
        c = relu_frag(c);                    // delta
        int n = nt * 16 + ncol;
#pragma unroll
        for (int r = 0; r < 8; ++r) {
            int m = mbase + r;
            int s = sA[m], d = dA[m];
            long long e = e0 + m;
            float w   = ex[e * 64 + n] * invden[(long long)d * 64 + n];
            float msg = w * (v[(long long)s * 64 + n] + c[r]);
            atomicAdd(acc + (long long)d * 64 + n, msg);
        }
    }
}

// ---------------------------------------------------------------------------
// Kernel 5: out = relu(acc @ W_out^T + b_out)
// ---------------------------------------------------------------------------
__global__ void k_node_out(const float* __restrict__ acc,
                           const __bf16* __restrict__ Wp_out, const float* __restrict__ b_out,
                           float* __restrict__ out, int ntiles) {
    __shared__ __align__(16) __bf16 tiles[4][16 * 64];
    int wid  = threadIdx.x >> 5;
    int tile = blockIdx.x * 4 + wid;
    if (tile >= ntiles) return;
    __bf16* t = tiles[wid];
    int lane  = threadIdx.x & 31;
    long long row0 = (long long)tile * 16;

    float bo[4];
#pragma unroll
    for (int nt = 0; nt < 4; ++nt) bo[nt] = b_out[nt * 16 + (lane & 15)];

    {
        int r  = lane >> 1;
        int c0 = (lane & 1) * 32;
        const float4* src = (const float4*)(acc + (row0 + r) * 64 + c0);
        v8bf* dst = (v8bf*)(t + r * 64 + c0);
        dst[0] = cvt8(src[0], src[1]);
        dst[1] = cvt8(src[2], src[3]);
        dst[2] = cvt8(src[4], src[5]);
        dst[3] = cvt8(src[6], src[7]);
    }
    __builtin_amdgcn_wave_barrier();
    v16bf a0 = load_A_lds(t, 64, 0);
    v16bf a1 = load_A_lds(t, 64, 1);
#pragma unroll
    for (int nt = 0; nt < 4; ++nt) {
        v8f c = splat_frag(bo[nt]);
        c = wmma_bf16(a0, load_B_frag(Wp_out, nt * 2 + 0), c);
        c = wmma_bf16(a1, load_B_frag(Wp_out, nt * 2 + 1), c);
        store_C_global(out, row0, nt, relu_frag(c));
    }
}

// ---------------------------------------------------------------------------
extern "C" void kernel_launch(void* const* d_in, const int* in_sizes, int n_in,
                              void* d_out, int out_size, void* d_ws, size_t ws_size,
                              hipStream_t stream) {
    const float* x     = (const float*)d_in[0];
    const float* pos   = (const float*)d_in[1];
    const int*   ei    = (const int*)  d_in[2];
    const float* W_in  = (const float*)d_in[3];
    const float* b_in  = (const float*)d_in[4];
    const float* W_out = (const float*)d_in[5];
    const float* b_out = (const float*)d_in[6];
    const float* W_lin = (const float*)d_in[7];
    const float* W_src = (const float*)d_in[8];
    const float* W_dst = (const float*)d_in[9];
    const float* pw1   = (const float*)d_in[10];
    const float* pb1   = (const float*)d_in[11];
    const float* pw2   = (const float*)d_in[12];
    const float* pb2   = (const float*)d_in[13];
    const float* aw1   = (const float*)d_in[14];
    const float* ab1   = (const float*)d_in[15];
    const float* aw2   = (const float*)d_in[16];
    const float* ab2   = (const float*)d_in[17];

    const int N  = in_sizes[0] / 64;     // 50000 (divisible by 16)
    const int E  = in_sizes[2] / 2;      // 800000
    const int EN = E + N;                // 850000 (divisible by 16)

    const long long n64 = (long long)N * 64;
    float* ws    = (float*)d_ws;
    float* v     = ws;                   // N x 64
    float* ks    = ws + 1 * n64;         // N x 64
    float* qd    = ws + 2 * n64;         // N x 64
    float* amax  = ws + 3 * n64;         // N x 64  (zeroed)
    float* denom = ws + 4 * n64;         // N x 64  (zeroed; inverted in 3b)
    float* acc   = ws + 5 * n64;         // N x 64  (zeroed)
    float* alpha = ws + 6 * n64;         // EN x 64

    // packed bf16 weight-fragment tables after alpha buffer
    __bf16* wp    = (__bf16*)(alpha + (long long)EN * 64);
    __bf16* p_in  = wp + 0 * 4096;
    __bf16* p_lin = wp + 1 * 4096;
    __bf16* p_src = wp + 2 * 4096;
    __bf16* p_dst = wp + 3 * 4096;
    __bf16* p_aw1 = wp + 4 * 4096;
    __bf16* p_aw2 = wp + 5 * 4096;
    __bf16* p_pw2 = wp + 6 * 4096;
    __bf16* p_out = wp + 7 * 4096;
    __bf16* p_pw1 = wp + 8 * 4096;      // 2048 entries (K padded to 32)

    const int tilesN = (N + 15) / 16;
    const int tilesE = (EN + 15) / 16;

    { // zero amax/denom/acc (3*n64 contiguous floats, float4 stores)
        long long nz4 = (3 * n64) / 4;
        k_zero4<<<(int)((nz4 + 255) / 256), 256, 0, stream>>>((float4*)amax, nz4);
    }
    { // pack weights (8*4096 + 2048 entries)
        int total = 8 * 4096 + 2048;
        k_pack_weights<<<(total + 255) / 256, 256, 0, stream>>>(
            W_in, W_lin, W_src, W_dst, aw1, aw2, pw2, W_out, pw1, wp);
    }
    k_node_proj<<<(tilesN + 3) / 4, 128, 0, stream>>>(
        x, p_in, b_in, p_lin, p_src, p_dst, v, ks, qd, tilesN);
    k_edge_alpha<<<(tilesE + 3) / 4, 128, 0, stream>>>(
        ei, ks, qd, p_aw1, ab1, p_aw2, ab2, alpha, amax, E, tilesE);
    {
        long long total4 = (long long)EN * 16;
        k_softmax_norm<<<(int)((total4 + 255) / 256), 256, 0, stream>>>(
            ei, amax, alpha, denom, E, total4);
    }
    {
        long long n4 = n64 / 4;
        k_inv_denom<<<(int)((n4 + 255) / 256), 256, 0, stream>>>((float4*)denom, n4);
    }
    k_edge_msg<<<(tilesE + 3) / 4, 128, 0, stream>>>(
        ei, pos, v, p_pw1, pb1, p_pw2, pb2, alpha, denom, acc, E, tilesE);
    k_node_out<<<(tilesN + 3) / 4, 128, 0, stream>>>(
        acc, p_out, b_out, (float*)d_out, tilesN);
}
